// LIT_8985071583515
// MI455X (gfx1250) — matrix operand order repaired and checked by reading, not compiled, not run on hardware
//
#include <hip/hip_runtime.h>

// ---------------- problem constants ----------------
#define HF 64
#define WF 64
#define HWF 4096
#define BB 2
#define QQ 2304
#define NROWS (BB * QQ)      // 4608
#define DIMC 192
#define HEADS 8
#define DKH 24
#define CFN 64
#define RA 49
#define INDIM 15682
#define KP 15712             // 491 * 32, zero-padded K
#define KTILES (KP / 32)     // 491
#define HID 256

typedef __attribute__((ext_vector_type(16))) __bf16 bf16x16;
typedef __attribute__((ext_vector_type(8)))  float  floatx8;
typedef unsigned int u32x4 __attribute__((ext_vector_type(4)));
typedef int          i32x8 __attribute__((ext_vector_type(8)));
typedef int          i32x4 __attribute__((ext_vector_type(4)));

// ---- Tensor Data Mover: 2D tile of bf16 (data_size=2B) global -> LDS ----
// D# layout per CDNA5 ISA ch.8 (group0: count/lds/global/type, group1: dims).
// This toolchain's builtin is the 6-arg clang-23 form:
//   (u32x4 g0, i32x8 g1, i32x4 g2, i32x4 g3, i32x8 aux, i32 cpol)
__device__ __forceinline__ void tdm_load_2d_bf16(unsigned lds_addr,
                                                 const __bf16* gptr,
                                                 unsigned tile_d0,     // elems per row
                                                 unsigned tile_d1,     // rows
                                                 unsigned stride_el) { // row stride, elems
    unsigned long long ga = (unsigned long long)(uintptr_t)gptr;
    u32x4 g0;
    g0[0] = 1u;                                            // count=1, user mode
    g0[1] = lds_addr;                                      // LDS byte address
    g0[2] = (unsigned)(ga & 0xffffffffu);                  // global_addr[31:0]
    g0[3] = (unsigned)((ga >> 32) & 0x01ffffffu)           // global_addr[56:32]
          | 0x80000000u;                                   // type=2 ("image")
    i32x8 g1;
    g1[0] = (int)(1u << 16);                               // wg_mask=0, data_size=1 (2B)
    g1[1] = (int)((stride_el & 0xffffu) << 16);            // tensor_dim0[15:0]
    g1[2] = (int)((stride_el >> 16) & 0xffffu)             // tensor_dim0[31:16]
          | (int)((tile_d1 & 0xffffu) << 16);              // tensor_dim1[15:0]
    g1[3] = (int)((tile_d0 & 0xffffu) << 16);              // tensor_dim1[31:16]=0, tile_dim0
    g1[4] = (int)(tile_d1 & 0xffffu);                      // tile_dim1, tile_dim2=0
    g1[5] = (int)stride_el;                                // tensor_dim0_stride[31:0]
    g1[6] = 0;                                             // stride hi + dim1_stride lo
    g1[7] = 0;
    i32x4 gz4 = {0, 0, 0, 0};
    i32x8 gz8 = {0, 0, 0, 0, 0, 0, 0, 0};
    __builtin_amdgcn_tensor_load_to_lds(g0, g1, gz4, gz4, gz8, 0);
}

// ---------------- kernel 1: encoder conv 3->64, 3x3 SAME ----------------
__global__ __launch_bounds__(256) void k_enc(const float* __restrict__ inp,
                                             const float* __restrict__ w,
                                             const float* __restrict__ bias,
                                             float* __restrict__ feat) {
    int gid = blockIdx.x * 256 + threadIdx.x;
    if (gid >= BB * 64 * HWF) return;
    int x = gid & 63, y = (gid >> 6) & 63, oc = (gid >> 12) & 63, b = gid >> 18;
    float acc = bias[oc];
    for (int ic = 0; ic < 3; ++ic)
        for (int ky = 0; ky < 3; ++ky) {
            int iy = y + ky - 1;
            if (iy < 0 || iy >= HF) continue;
            for (int kx = 0; kx < 3; ++kx) {
                int ix = x + kx - 1;
                if (ix < 0 || ix >= WF) continue;
                acc += inp[((b * 3 + ic) * HF + iy) * WF + ix] *
                       w[((oc * 3 + ic) * 3 + ky) * 3 + kx];
            }
        }
    feat[((b * 64 + oc) * HF + y) * WF + x] = acc;
}

// ---------------- kernel 2: fused q/k/v/f convs 64->{192,192,192,64} ----------------
__global__ __launch_bounds__(256) void k_qkvf(const float* __restrict__ feat,
                                              const float* __restrict__ qw, const float* __restrict__ qb,
                                              const float* __restrict__ kw, const float* __restrict__ kb,
                                              const float* __restrict__ vw, const float* __restrict__ vb,
                                              const float* __restrict__ fw, const float* __restrict__ fb,
                                              float* __restrict__ fq, float* __restrict__ fk,
                                              float* __restrict__ fv, float* __restrict__ ff) {
    int gid = blockIdx.x * 256 + threadIdx.x;
    if (gid >= BB * 640 * HWF) return;
    int pix = gid & 4095, tmp = gid >> 12;
    int oc = tmp % 640, b = tmp / 640;
    int y = pix >> 6, x = pix & 63;
    const float* w; const float* bs; float* out; int ocl; int C;
    if (oc < 192)      { w = qw; bs = qb; out = fq; ocl = oc;       C = 192; }
    else if (oc < 384) { w = kw; bs = kb; out = fk; ocl = oc - 192; C = 192; }
    else if (oc < 576) { w = vw; bs = vb; out = fv; ocl = oc - 384; C = 192; }
    else               { w = fw; bs = fb; out = ff; ocl = oc - 576; C = 64;  }
    float acc = bs[ocl];
    for (int ic = 0; ic < 64; ++ic)
        for (int ky = 0; ky < 3; ++ky) {
            int iy = y + ky - 1;
            if (iy < 0 || iy >= HF) continue;
            for (int kx = 0; kx < 3; ++kx) {
                int ix = x + kx - 1;
                if (ix < 0 || ix >= WF) continue;
                acc += feat[((b * 64 + ic) * HF + iy) * WF + ix] *
                       w[((ocl * 64 + ic) * 3 + ky) * 3 + kx];
            }
        }
    out[((size_t)b * C + ocl) * HWF + pix] = acc;
}

// ---------------- kernel 3: w1 fp32 -> bf16, K padded to KP ----------------
__global__ __launch_bounds__(256) void k_w1(const float* __restrict__ w1,
                                            __bf16* __restrict__ w1b) {
    int e = blockIdx.x * 256 + threadIdx.x;
    if (e >= KP * 256) return;
    int k = e >> 8, n = e & 255;
    float v = (k < INDIM) ? w1[k * 256 + n] : 0.0f;
    w1b[e] = (__bf16)v;
}

// ---------------- kernel 4: attention (sim + positional bias + softmax) ----------------
__global__ __launch_bounds__(64) void k_attn(const float* __restrict__ coord,
                                             const float* __restrict__ fq_,
                                             const float* __restrict__ fk_,
                                             const float* __restrict__ pbw,
                                             const float* __restrict__ pbb,
                                             float* __restrict__ attnb,
                                             float* __restrict__ dout) {
    int bq = blockIdx.x;
    int b = bq / QQ, q = bq % QQ;
    int t = threadIdx.x;
    __shared__ float qsh[DIMC];
    __shared__ float simsh[RA][HEADS];
    __shared__ float mx[HEADS], sinv[HEADS];

    float cy = coord[bq * 2 + 0], cx = coord[bq * 2 + 1];
    float ccy = fminf(fmaxf(cy, -1.0f + 1e-6f), 1.0f - 1e-6f);
    float ccx = fminf(fmaxf(cx, -1.0f + 1e-6f), 1.0f - 1e-6f);
    float py = ((ccy + 1.0f) * HF - 1.0f) * 0.5f;
    float px = ((ccx + 1.0f) * WF - 1.0f) * 0.5f;

    // q_: bilinear gather with zero padding
    int y0 = (int)floorf(py), x0 = (int)floorf(px);
    float wy = py - (float)y0, wx = px - (float)x0;
    for (int c = t; c < DIMC; c += 64) {
        float acc = 0.0f;
        const float* fc = fq_ + ((size_t)b * DIMC + c) * HWF;
        #pragma unroll
        for (int cn = 0; cn < 4; ++cn) {
            int yy = y0 + (cn >> 1), xx = x0 + (cn & 1);
            float wgt = ((cn >> 1) ? wy : (1.0f - wy)) * ((cn & 1) ? wx : (1.0f - wx));
            if (yy >= 0 && yy < HF && xx >= 0 && xx < WF) acc += fc[yy * WF + xx] * wgt;
        }
        qsh[c] = acc;
    }
    int iy0 = (int)rintf(py), ix0 = (int)rintf(px);
    __syncthreads();

    if (t < RA) {
        int dy = t / 7 - 3, dx = t % 7 - 3;
        int sy = iy0 + dy, sx = ix0 + dx;
        bool valid = (sy >= 0 && sy < HF && sx >= 0 && sx < WF);
        int pix = valid ? (sy * WF + sx) : 0;
        float cky = -1.0f + (2.0f * (float)(iy0 + dy) + 1.0f) * (1.0f / HF);
        float ckx = -1.0f + (2.0f * (float)(ix0 + dx) + 1.0f) * (1.0f / WF);
        float ry = (cy - cky) * HF, rx = (cx - ckx) * WF;
        const float invsq = 0.20412414523193154f;   // 1/sqrt(24)
        for (int h = 0; h < HEADS; ++h) {
            float s = 0.0f;
            if (valid) {
                const float* fkb = fk_ + ((size_t)b * DIMC + h * DKH) * HWF + pix;
                for (int dd = 0; dd < DKH; ++dd)
                    s += qsh[h * DKH + dd] * fkb[(size_t)dd * HWF];
            }
            simsh[t][h] = s * invsq + ry * pbw[h] + rx * pbw[HEADS + h] + pbb[h];
        }
    }
    __syncthreads();
    if (t < HEADS) {
        float m = -1e30f;
        for (int s = 0; s < RA; ++s) m = fmaxf(m, simsh[s][t]);
        float sum = 0.0f;
        for (int s = 0; s < RA; ++s) sum += expf(simsh[s][t] - m);
        mx[t] = m; sinv[t] = 1.0f / sum;
    }
    __syncthreads();
    if (t < RA) {
        for (int h = 0; h < HEADS; ++h) {
            float a = expf(simsh[t][h] - mx[h]) * sinv[h];
            attnb[((size_t)bq * RA + t) * HEADS + h] = a;
            if (b == 0 && q == 10 && h == 0) dout[NROWS * 3 + t] = a;  // attn_map tail
        }
    }
}

// ---------------- kernel 5: activation rows (v-weighted + Fourier PE) in bf16 ----------------
__global__ __launch_bounds__(256) void k_act(const float* __restrict__ coord,
                                             const float* __restrict__ cell,
                                             const float* __restrict__ fv_,
                                             const float* __restrict__ ff_,
                                             const float* __restrict__ pew,
                                             const float* __restrict__ attnb,
                                             __bf16* __restrict__ actA) {
    int bq = blockIdx.x;
    int b = bq / QQ;
    int t = threadIdx.x;
    __shared__ float fr[RA][CFN];
    __shared__ float fcr[RA][CFN];
    __shared__ float fci[RA][CFN];
    __shared__ float ash[RA][HEADS];
    __shared__ float rely[RA], relx[RA];

    float cy = coord[bq * 2 + 0], cx = coord[bq * 2 + 1];
    float ccy = fminf(fmaxf(cy, -1.0f + 1e-6f), 1.0f - 1e-6f);
    float ccx = fminf(fmaxf(cx, -1.0f + 1e-6f), 1.0f - 1e-6f);
    int iy0 = (int)rintf(((ccy + 1.0f) * HF - 1.0f) * 0.5f);
    int ix0 = (int)rintf(((ccx + 1.0f) * WF - 1.0f) * 0.5f);

    for (int e = t; e < RA * HEADS; e += 256)
        ash[e / HEADS][e % HEADS] = attnb[(size_t)bq * RA * HEADS + e];
    if (t < RA) {
        int dy = t / 7 - 3, dx = t % 7 - 3;
        float cky = -1.0f + (2.0f * (float)(iy0 + dy) + 1.0f) * (1.0f / HF);
        float ckx = -1.0f + (2.0f * (float)(ix0 + dx) + 1.0f) * (1.0f / WF);
        rely[t] = (cy - cky) * HF;
        relx[t] = (cx - ckx) * WF;
    }
    for (int e = t; e < RA * CFN; e += 256) {
        int s = e >> 6, c = e & 63;
        int sy = iy0 + s / 7 - 3, sx = ix0 + s % 7 - 3;
        bool valid = (sy >= 0 && sy < HF && sx >= 0 && sx < WF);
        fr[s][c] = valid ? ff_[((size_t)b * CFN + c) * HWF + sy * WF + sx] : 0.0f;
    }
    __syncthreads();

    // 7x7 2D DFT (ortho) per channel
    float c7[7], s7[7];
    #pragma unroll
    for (int k = 0; k < 7; ++k) {
        float ang = -6.283185307179586f * (float)k / 7.0f;
        sincosf(ang, &s7[k], &c7[k]);
    }
    for (int e = t; e < RA * CFN; e += 256) {
        int so = e >> 6, c = e & 63;
        int u = so / 7, v = so % 7;
        float re = 0.0f, im = 0.0f;
        for (int a = 0; a < 7; ++a) {
            int ua = u * a;
            for (int bb2 = 0; bb2 < 7; ++bb2) {
                int k = (ua + v * bb2) % 7;
                float f = fr[a * 7 + bb2][c];
                re += f * c7[k];
                im += f * s7[k];
            }
        }
        fcr[so][c] = re * (1.0f / 7.0f);
        fci[so][c] = im * (1.0f / 7.0f);
    }
    __syncthreads();

    const size_t rowbase = (size_t)bq * KP;
    // v_weighted -> columns [s*320 .. s*320+191]
    for (int e = t; e < RA * DIMC; e += 256) {
        int s = e / DIMC, d = e % DIMC;
        int sy = iy0 + s / 7 - 3, sx = ix0 + s % 7 - 3;
        bool valid = (sy >= 0 && sy < HF && sx >= 0 && sx < WF);
        float vv = valid ? fv_[((size_t)b * DIMC + d) * HWF + sy * WF + sx] : 0.0f;
        actA[rowbase + s * 320 + d] = (__bf16)(vv * ash[s][d / DKH]);
    }
    // enc -> columns [s*320+192 .. s*320+319]
    for (int e = t; e < RA * CFN; e += 256) {
        int s = e >> 6, cf = e & 63;
        float ph = rely[s] * pew[cf] + relx[s] * pew[CFN + cf];
        float sn, cs;
        sincosf(ph, &sn, &cs);
        float re = fcr[s][cf], im = fci[s][cf];
        actA[rowbase + s * 320 + 192 + 2 * cf + 0] = (__bf16)(cs * re - sn * im);
        actA[rowbase + s * 320 + 192 + 2 * cf + 1] = (__bf16)(cs * im + sn * re);
    }
    if (t == 0) {
        actA[rowbase + 15680] = (__bf16)(cell[bq * 2 + 0] * HF);
        actA[rowbase + 15681] = (__bf16)(cell[bq * 2 + 1] * WF);
    }
    if (t < KP - INDIM) actA[rowbase + INDIM + t] = (__bf16)0.0f;
}

// ---------------- kernel 6: WMMA bf16 GEMM with TDM staging ----------------
// Block: 32 rows x 256 cols. 8 waves, each wave computes a 2x2 grid of 16x16
// tiles. A (32x32) and B (32x256) tiles staged in LDS by the Tensor Data
// Mover (wave 0 issues, s_wait_tensorcnt, workgroup barrier releases).
__global__ __launch_bounds__(256) void k_gemm(const __bf16* __restrict__ actA,
                                              const __bf16* __restrict__ w1b,
                                              const float* __restrict__ b1,
                                              float* __restrict__ hbuf) {
    __shared__ __bf16 Ash[32][32];
    __shared__ __bf16 Bsh[32][256];
    int m0 = blockIdx.x * 32;
    int t = threadIdx.x;
    int w = t >> 5, lane = t & 31;
    int laneHalf = lane >> 4, mrow = lane & 15;

    floatx8 acc00 = {}, acc01 = {}, acc10 = {}, acc11 = {};
    unsigned ldsA = (unsigned)(uintptr_t)(&Ash[0][0]);
    unsigned ldsB = (unsigned)(uintptr_t)(&Bsh[0][0]);

    for (int kt = 0; kt < KTILES; ++kt) {
        int k0 = kt * 32;
        if (w == 0) {
            // A tile: 32 rows x 32 bf16, row stride KP elements
            tdm_load_2d_bf16(ldsA, actA + (size_t)m0 * KP + k0, 32u, 32u, (unsigned)KP);
            // B tile: 32 contiguous rows of 256 bf16 = one 8192-elem line
            tdm_load_2d_bf16(ldsB, w1b + (size_t)k0 * 256, 8192u, 1u, 8192u);
            __builtin_amdgcn_s_wait_tensorcnt(0);
        }
        __syncthreads();

        // pack fragments per documented CDNA5 16-bit WMMA layouts
        bf16x16 af0, af1, bf0, bf1;
        #pragma unroll
        for (int j = 0; j < 16; ++j) {
            int kk = ((j >= 8) ? 16 : 0) + laneHalf * 8 + ((j >> 1) & 3) * 2 + (j & 1);
            af0[j] = Ash[mrow][kk];
            af1[j] = Ash[16 + mrow][kk];
        }
        int ncol0 = w * 32 + mrow, ncol1 = w * 32 + 16 + mrow;
        #pragma unroll
        for (int j = 0; j < 16; ++j) {
            int kk = ((j >= 8) ? 16 : 0) + laneHalf * 8 + ((j >> 1) & 3) * 2 + (j & 1);
            bf0[j] = Bsh[kk][ncol0];
            bf1[j] = Bsh[kk][ncol1];
        }
        acc00 = __builtin_amdgcn_wmma_f32_16x16x32_bf16(false, af0, false, bf0,
                                                        (short)0, acc00, false, false);
        acc01 = __builtin_amdgcn_wmma_f32_16x16x32_bf16(false, af0, false, bf1,
                                                        (short)0, acc01, false, false);
        acc10 = __builtin_amdgcn_wmma_f32_16x16x32_bf16(false, af1, false, bf0,
                                                        (short)0, acc10, false, false);
        acc11 = __builtin_amdgcn_wmma_f32_16x16x32_bf16(false, af1, false, bf1,
                                                        (short)0, acc11, false, false);
        __syncthreads();
    }

    #pragma unroll
    for (int r = 0; r < 8; ++r) {
        int row0 = m0 + r + 8 * laneHalf;
        int row1 = row0 + 16;
        int c0 = w * 32 + mrow;
        int c1 = c0 + 16;
        hbuf[(size_t)row0 * HID + c0] = fmaxf(acc00[r] + b1[c0], 0.0f);
        hbuf[(size_t)row0 * HID + c1] = fmaxf(acc01[r] + b1[c1], 0.0f);
        hbuf[(size_t)row1 * HID + c0] = fmaxf(acc10[r] + b1[c0], 0.0f);
        hbuf[(size_t)row1 * HID + c1] = fmaxf(acc11[r] + b1[c1], 0.0f);
    }
}

// ---------------- kernel 7: layer2 + border-bilinear residual ----------------
__global__ __launch_bounds__(256) void k_out(const float* __restrict__ hbuf,
                                             const float* __restrict__ w2,
                                             const float* __restrict__ b2,
                                             const float* __restrict__ inp,
                                             const float* __restrict__ coord,
                                             float* __restrict__ dout) {
    int e = blockIdx.x * 256 + threadIdx.x;
    if (e >= NROWS * 3) return;
    int row = e / 3, ch = e % 3;
    int b = row / QQ;
    float acc = b2[ch];
    const float* hr = hbuf + (size_t)row * HID;
    for (int j = 0; j < HID; ++j) acc += hr[j] * w2[j * 3 + ch];

    float cy = coord[row * 2 + 0], cx = coord[row * 2 + 1];
    float py = fminf(fmaxf(((cy + 1.0f) * HF - 1.0f) * 0.5f, 0.0f), (float)(HF - 1));
    float px = fminf(fmaxf(((cx + 1.0f) * WF - 1.0f) * 0.5f, 0.0f), (float)(WF - 1));
    int y0 = (int)floorf(py), x0 = (int)floorf(px);
    float wy = py - (float)y0, wx = px - (float)x0;
    int y1 = min(y0 + 1, HF - 1), x1 = min(x0 + 1, WF - 1);
    const float* ip = inp + ((size_t)b * 3 + ch) * HWF;
    float res = ip[y0 * WF + x0] * (1.0f - wy) * (1.0f - wx)
              + ip[y0 * WF + x1] * (1.0f - wy) * wx
              + ip[y1 * WF + x0] * wy * (1.0f - wx)
              + ip[y1 * WF + x1] * wy * wx;
    dout[e] = acc + res;
}

// ---------------- launch ----------------
extern "C" void kernel_launch(void* const* d_in, const int* in_sizes, int n_in,
                              void* d_out, int out_size, void* d_ws, size_t ws_size,
                              hipStream_t stream) {
    const float* inp   = (const float*)d_in[0];
    const float* coord = (const float*)d_in[1];
    const float* cell  = (const float*)d_in[2];
    const float* enc_w = (const float*)d_in[3];
    const float* enc_b = (const float*)d_in[4];
    const float* qw    = (const float*)d_in[5];
    const float* qb    = (const float*)d_in[6];
    const float* kw    = (const float*)d_in[7];
    const float* kb    = (const float*)d_in[8];
    const float* vw    = (const float*)d_in[9];
    const float* vb    = (const float*)d_in[10];
    const float* fw    = (const float*)d_in[11];
    const float* fb    = (const float*)d_in[12];
    const float* pbw   = (const float*)d_in[13];
    const float* pbb   = (const float*)d_in[14];
    const float* pew   = (const float*)d_in[15];
    const float* w1    = (const float*)d_in[16];
    const float* b1    = (const float*)d_in[17];
    const float* w2    = (const float*)d_in[18];
    const float* b2    = (const float*)d_in[19];
    float* dout = (float*)d_out;

    char* ws = (char*)d_ws;
    size_t off = 0;
    auto take = [&](size_t bytes) -> void* {
        void* p = ws + off;
        off = (off + bytes + 255) & ~(size_t)255;
        return p;
    };
    float*  feat  = (float*) take((size_t)BB * 64 * HWF * 4);
    float*  fq    = (float*) take((size_t)BB * DIMC * HWF * 4);
    float*  fk    = (float*) take((size_t)BB * DIMC * HWF * 4);
    float*  fv    = (float*) take((size_t)BB * DIMC * HWF * 4);
    float*  ff    = (float*) take((size_t)BB * CFN * HWF * 4);
    float*  attnb = (float*) take((size_t)NROWS * RA * HEADS * 4);
    __bf16* w1b   = (__bf16*)take((size_t)KP * HID * 2);
    __bf16* actA  = (__bf16*)take((size_t)NROWS * KP * 2);
    float*  hbuf  = (float*) take((size_t)NROWS * HID * 4);

    k_enc <<<(BB * 64 * HWF + 255) / 256, 256, 0, stream>>>(inp, enc_w, enc_b, feat);
    k_qkvf<<<(BB * 640 * HWF + 255) / 256, 256, 0, stream>>>(feat, qw, qb, kw, kb,
                                                             vw, vb, fw, fb,
                                                             fq, fk, fv, ff);
    k_w1  <<<(KP * HID + 255) / 256, 256, 0, stream>>>(w1, w1b);
    k_attn<<<NROWS, 64, 0, stream>>>(coord, fq, fk, pbw, pbb, attnb, dout);
    k_act <<<NROWS, 256, 0, stream>>>(coord, cell, fv, ff, pew, attnb, actA);
    k_gemm<<<NROWS / 32, 256, 0, stream>>>(actA, w1b, b1, hbuf);
    k_out <<<(NROWS * 3 + 255) / 256, 256, 0, stream>>>(hbuf, w2, b2, inp, coord, dout);
}
